// LSTM_11879879541009
// MI455X (gfx1250) — matrix operand order repaired and checked by reading, not compiled, not run on hardware
//
#include <hip/hip_runtime.h>
#include <hip/hip_bf16.h>
#include <cstdint>

#define SEQ   512
#define BATCH 64
#define INSZ  1024
#define HID   1024
#define NG    4096   // 4 gates * HID (P row stride; layout is [row][n*4+g])

#define LDS_ROW 1032           // padded row stride (elements): 2064 B == 4 dwords mod 64 banks

typedef __attribute__((ext_vector_type(16))) __bf16 v16bf;
typedef __attribute__((ext_vector_type(8)))  float  v8f;

struct F { union { v16bf v; uint4 q[2]; }; };

__device__ __forceinline__ unsigned short f2bf(float f) {
    unsigned int u = __float_as_uint(f);
    u += 0x7FFFu + ((u >> 16) & 1u);          // round-to-nearest-even
    return (unsigned short)(u >> 16);
}
__device__ __forceinline__ float sigmoid_(float x) { return 1.0f / (1.0f + __expf(-x)); }
__device__ __forceinline__ float tanh_(float x)    { return 1.0f - 2.0f / (__expf(2.0f * x) + 1.0f); }

// Async-stage a 16-row x 1024-col bf16 tile (32 KB) from global into LDS.
// 256 threads x 8 issues x 16 B = 32 KB.  Tracked by ASYNCcnt; barrier before use.
__device__ __forceinline__ void stage_A_async(unsigned short* sA,
                                              const unsigned short* __restrict__ src,
                                              int tid) {
    const uint32_t sbase = (uint32_t)(uintptr_t)sA;   // LDS aperture is 4GB-aligned -> low 32b = LDS offset
#pragma unroll
    for (int r8 = 0; r8 < 8; ++r8) {
        int i   = (r8 << 8) + tid;     // 0..2047
        int row = i >> 7;              // 0..15   (128 x 16B chunks per 2048B row)
        int ch  = i & 127;
        uint64_t ga = (uint64_t)(uintptr_t)(src + ((size_t)row << 10)) + ((uint32_t)ch << 4);
        uint32_t la = sbase + (uint32_t)row * (LDS_ROW * 2u) + ((uint32_t)ch << 4);
        asm volatile("global_load_async_to_lds_b128 %0, %1, off"
                     :: "v"(la), "v"(ga) : "memory");
    }
    asm volatile("s_wait_asynccnt 0x0" ::: "memory");
    __syncthreads();
}

// Load one K-step fragment set: A (2x ds_load_b128) + 4 gate B frags (8x global_load_b128).
__device__ __forceinline__ void load_frag_set(F& a, F (&b)[4],
                                              const unsigned short* aRow,
                                              const unsigned short* const (&bBase)[4],
                                              int k0) {
    a.q[0] = *(const uint4*)(aRow + k0);
    a.q[1] = *(const uint4*)(aRow + k0 + 8);
#pragma unroll
    for (int g = 0; g < 4; ++g) {
        b[g].q[0] = *(const uint4*)(bBase[g] + k0);
        b[g].q[1] = *(const uint4*)(bBase[g] + k0 + 8);
    }
}

// Generation-counting grid barrier (all blocks co-resident: grid is 32 blocks).
// Release: every thread fences its own stores before arrival.
// Acquire: every thread fences (invalidates near caches) after exit.
__device__ __forceinline__ void grid_sync(unsigned int* cnt, unsigned int target) {
    __threadfence();                           // release this thread's hout/out stores
    __syncthreads();
    if (threadIdx.x == 0) {
        atomicAdd(cnt, 1u);
        while (atomicAdd(cnt, 0u) < target)
            __builtin_amdgcn_s_sleep(1);
    }
    __syncthreads();
    __threadfence();                           // acquire: invalidate stale near caches
}

// ---- Phase 2 (defined first so it leads the disasm): persistent kernel runs all 512
//      timesteps with a grid barrier per step.  Cell state c lives in registers for the
//      whole sequence.  One 8-wave block per WGP; waves_per_eu(4,4) -> 256-VGPR budget:
//      enough for ~4 K-iterations of loads in flight, no VGPR-MSB juggling, no spills. ----
__global__ void __launch_bounds__(256) __attribute__((amdgpu_waves_per_eu(4, 4)))
lstm_seq_kernel(const unsigned short* __restrict__ Wt,   // [4][1024][2048] bf16
                const float*          __restrict__ P,    // [32768][1024][4] fp32 (gate-interleaved)
                unsigned short*       __restrict__ hb0,  // [64][1024] bf16 ping
                unsigned short*       __restrict__ hb1,  // [64][1024] bf16 pong
                float*                __restrict__ out,  // [512][64][1024] fp32
                unsigned int*         __restrict__ bar)
{
    __shared__ unsigned short sA[16 * LDS_ROW];

    const int wid  = (blockIdx.x << 3) | (threadIdx.x >> 5);   // 0..255
    const int lane = threadIdx.x & 31;
    const int mt = wid >> 6;                 // 0..3  : batch tile (same for all 8 waves in block)
    const int nt = wid & 63;                 // 0..63 : hidden tile
    const int m0 = mt << 4;
    const int n0 = nt << 4;

    const int koffA = (lane >> 4) << 4;
    const int bn    = lane & 15;
    const int kbB   = (lane >> 4) << 4;
    const int nabs  = n0 + bn;
    const int mbase = m0 + ((lane >> 4) << 3);

    const unsigned short* aRow = sA + (lane & 15) * LDS_ROW + koffA;
    const unsigned short* bBase[4];
#pragma unroll
    for (int g = 0; g < 4; ++g)
        bBase[g] = Wt + ((size_t)g << 21) + ((size_t)nabs << 11) + kbB;        // h-part: k 0..1023

    v8f creg = {};                           // cell state: 8 rows per lane, registers for all 512 steps
    unsigned short* hbuf[2] = { hb0, hb1 };

    for (int t = 0; t < SEQ; ++t) {
        const unsigned short* hin  = hbuf[t & 1];
        unsigned short*       hout = hbuf[(t & 1) ^ 1];

        stage_A_async(sA, hin + ((size_t)m0 << 10), threadIdx.x);

        // hoist the P preactivation stream (HBM): 8 coalesced float4 loads per lane,
        // all four gates per load (gate-interleaved layout)
        const float* Pt = P + (size_t)t * (BATCH * NG);
        float4 pv[8];
#pragma unroll
        for (int r = 0; r < 8; ++r)
            pv[r] = *(const float4*)(Pt + ((size_t)(mbase + r) << 12) + ((size_t)nabs << 2));

        v8f acc[4] = {};
#pragma unroll
        for (int kk = 0; kk < 32; ++kk) {
            const int k0 = kk << 5;
            F a, b[4];
            load_frag_set(a, b, aRow, bBase, k0);
            __builtin_prefetch(bBase[0] + k0 + 512, 0, 0);
#pragma unroll
            for (int g = 0; g < 4; ++g)
                acc[g] = __builtin_amdgcn_wmma_f32_16x16x32_bf16(
                    false, a.v, false, b[g].v, (short)0, acc[g], false, false);
        }

#pragma unroll
        for (int r = 0; r < 8; ++r) {
            float fg = sigmoid_(acc[0][r] + pv[r].x);
            float ig = sigmoid_(acc[1][r] + pv[r].y);
            float og = sigmoid_(acc[2][r] + pv[r].z);
            float ch = tanh_  (acc[3][r] + pv[r].w);
            float cn = creg[r] * fg + ch * ig;
            creg[r]  = cn;
            float h  = tanh_(cn) * og;
            const size_t sidx = (size_t)(mbase + r) * HID + nabs;
            out[(size_t)t * (BATCH * HID) + sidx] = h;
            hout[sidx] = f2bf(h);
        }

        grid_sync(bar, (unsigned)(t + 1) * gridDim.x);
    }
}

// ---- Prep: W[g][k][n] fp32  ->  Wt[g][n][k] bf16 (transposed, K contiguous per row) ----
__global__ void conv_w_kernel(const float* __restrict__ Wf, const float* __restrict__ Wi,
                              const float* __restrict__ Wo, const float* __restrict__ Wc,
                              unsigned short* __restrict__ Wt) {
    int idx = blockIdx.x * blockDim.x + threadIdx.x;   // 4 * 2048 * 1024 threads
    int g   = idx >> 21;
    int rem = idx & ((1 << 21) - 1);
    int k   = rem >> 10;          // 0..2047 (z dim: 0..1023 = h rows, 1024..2047 = x rows)
    int n   = rem & 1023;         // output column
    const float* W = (g == 0) ? Wf : (g == 1) ? Wi : (g == 2) ? Wo : Wc;
    Wt[((size_t)g << 21) + ((size_t)n << 11) + k] = f2bf(W[rem]);
}

__global__ void conv_x_kernel(const float* __restrict__ x, unsigned short* __restrict__ xb, int n) {
    int i = blockIdx.x * blockDim.x + threadIdx.x;
    if (i < n) xb[i] = f2bf(x[i]);
}

__global__ void init_state_kernel(unsigned short* __restrict__ hb0, unsigned int* __restrict__ bar) {
    int i = blockIdx.x * blockDim.x + threadIdx.x;     // 64*1024 threads
    hb0[i] = 0;
    if (i == 0) bar[0] = 0u;
}

// ---- Phase 1: P[t*64+b][n*4+g] = x_t @ W_x(+bias), all timesteps in parallel ----
__global__ void __launch_bounds__(256)
xgemm_kernel(const unsigned short* __restrict__ xb,   // [32768][1024] bf16
             const unsigned short* __restrict__ Wt,   // [4][1024][2048] bf16
             const float* __restrict__ bf, const float* __restrict__ bi,
             const float* __restrict__ bo, const float* __restrict__ bc,
             float* __restrict__ P)                   // [32768][1024][4] fp32 (gate-interleaved)
{
    __shared__ unsigned short sA[16 * LDS_ROW];       // 33 KB padded A tile

    const int wid  = (blockIdx.x << 3) | (threadIdx.x >> 5);
    const int lane = threadIdx.x & 31;
    const int mt = wid >> 6;                 // 0..2047 : 16-row M tile (same for all 8 waves in block)
    const int nb = wid & 63;                 // 0..63   : 64-col N block
    const int m0 = mt << 4;
    const int n0 = nb << 6;

    stage_A_async(sA, xb + ((size_t)m0 << 10), threadIdx.x);

    const int koffA = (lane >> 4) << 4;      // A: 0 or 16 (half-wave K split)
    const int bn    = lane & 15;             // B: local column
    const int kbB   = (lane >> 4) << 4;      // B: 0 or 16 (half-wave K split)

    const unsigned short* aRow = sA + (lane & 15) * LDS_ROW + koffA;
    const unsigned short* bBase[4];
    float bias[4];
    int   gidx[4], nlidx[4];
#pragma unroll
    for (int j = 0; j < 4; ++j) {
        int nabs = n0 + (j << 4) + bn;
        int g    = nabs >> 10;
        int nl   = nabs & 1023;
        const float* bp = (g == 0) ? bf : (g == 1) ? bi : (g == 2) ? bo : bc;
        bias[j]  = bp[nl];
        gidx[j]  = g;
        nlidx[j] = nl;
        bBase[j] = Wt + ((size_t)g << 21) + ((size_t)nl << 11) + INSZ + kbB;   // x-part: k 1024..2047
    }

    v8f acc[4] = {};
    F a[2], b[2][4];
    load_frag_set(a[0], b[0], aRow, bBase, 0);
#pragma unroll
    for (int kk = 0; kk < 32; ++kk) {
        const int cur = kk & 1;
        if (kk < 31)
            load_frag_set(a[cur ^ 1], b[cur ^ 1], aRow, bBase, (kk + 1) << 5);
#pragma unroll
        for (int j = 0; j < 4; ++j)
            acc[j] = __builtin_amdgcn_wmma_f32_16x16x32_bf16(
                false, a[cur].v, false, b[cur][j].v, (short)0, acc[j], false, false);
    }

    const int mbase = m0 + ((lane >> 4) << 3);
#pragma unroll
    for (int j = 0; j < 4; ++j) {
#pragma unroll
        for (int r = 0; r < 8; ++r)
            P[((size_t)(mbase + r) << 12) + ((size_t)nlidx[j] << 2) + gidx[j]] = acc[j][r] + bias[j];
    }
}

extern "C" void kernel_launch(void* const* d_in, const int* in_sizes, int n_in,
                              void* d_out, int out_size, void* d_ws, size_t ws_size,
                              hipStream_t stream) {
    (void)in_sizes; (void)n_in; (void)out_size; (void)ws_size;
    const float* x  = (const float*)d_in[0];
    const float* Wf = (const float*)d_in[1];
    const float* bf = (const float*)d_in[2];
    const float* Wi = (const float*)d_in[3];
    const float* bi = (const float*)d_in[4];
    const float* Wo = (const float*)d_in[5];
    const float* bo = (const float*)d_in[6];
    const float* Wc = (const float*)d_in[7];
    const float* bc = (const float*)d_in[8];
    float* out = (float*)d_out;

    char* ws = (char*)d_ws;
    unsigned short* Wt  = (unsigned short*)ws;                          // 16 MB  bf16 weights (transposed)
    unsigned short* xb  = (unsigned short*)(ws + ((size_t)16 << 20));   // 64 MB  bf16 x
    float*          P   = (float*)        (ws + ((size_t)80 << 20));    // 512 MB fp32 x-preactivations
    unsigned short* hb0 = (unsigned short*)(ws + ((size_t)592 << 20));  // 128 KB h ping (bf16)
    unsigned short* hb1 = hb0 + BATCH * HID;                            // 128 KB h pong (bf16)
    unsigned int*   bar = (unsigned int*)(hb1 + BATCH * HID);           // grid barrier counter

    conv_w_kernel<<<(4 * 2048 * 1024) / 256, 256, 0, stream>>>(Wf, Wi, Wo, Wc, Wt);
    conv_x_kernel<<<(SEQ * BATCH * INSZ) / 256, 256, 0, stream>>>(x, xb, SEQ * BATCH * INSZ);
    init_state_kernel<<<(BATCH * HID) / 256, 256, 0, stream>>>(hb0, bar);

    // 2048 M-tiles x 64 N-blocks = 131072 waves, 8 waves/block
    xgemm_kernel<<<16384, 256, 0, stream>>>(xb, Wt, bf, bi, bo, bc, P);

    // persistent recurrence: 32 blocks (co-resident, one per WGP), 512 steps, grid barrier per step
    lstm_seq_kernel<<<32, 256, 0, stream>>>(Wt, P, hb0, hb1, out, bar);
}
